// SimplifiedAttention_34875134443550
// MI455X (gfx1250) — matrix operand order repaired and checked by reading, not compiled, not run on hardware
//
#include <hip/hip_runtime.h>
#include <stdint.h>

// ---------------------------------------------------------------------------
// SimplifiedAttention for MI455X (gfx1250, wave32, WMMA + async-to-LDS)
//   qp = q Wq^T + bq ; kp = k Wk^T + bk ; vpT = (v Wv^T + bv)^T
//   S  = qp kp^T * (1/sqrt(64)) ; P = softmax(S) ; ctx = P vp
//   out = ctx Wo^T + bo
// All GEMMs: v_wmma_f32_16x16x32_bf16 (bf16 in, fp32 accumulate), LDS tiles
// filled with global_load_async_to_lds_b128 (ASYNCcnt), double-buffered.
// ---------------------------------------------------------------------------

typedef __bf16 bf16_t;
typedef __attribute__((ext_vector_type(16))) __bf16 v16bf;
typedef __attribute__((ext_vector_type(8)))  __bf16 v8bf;
typedef __attribute__((ext_vector_type(8)))  float  v8f;

#define D_MODEL 1024
#define SEQ     2048
#define BATCH   4

constexpr int BM = 128;   // block tile M
constexpr int BN = 128;   // block tile N
constexpr int BK = 64;    // block tile K (two WMMA K-steps)

// ---------------------------------------------------------------------------
// NT WMMA GEMM:  C = alpha * A * B^T + bias
//   A : bf16 [M][K] row-major (lda = K)
//   B : bf16 [N][K] row-major with row stride ldb
//   OUT_F32  : write fp32, else bf16
//   BIAS     : add fp32 bias[n]
//   OUT_TRANS: write C transposed, C[n][m], row stride ldcT (each lane then
//              stores 8 contiguous rows = one b128 store)
// grid = (N/BN, M/BM, batch) ; block = 256 threads = 8 waves (2x4),
// each wave computes a 64x32 region = 4x2 tiles of 16x16.
// ---------------------------------------------------------------------------
template <bool OUT_F32, bool BIAS, bool OUT_TRANS>
__global__ __launch_bounds__(256)
void gemm_wmma(const bf16_t* __restrict__ Ap, const bf16_t* __restrict__ Bp,
               const float* __restrict__ bias, void* __restrict__ Cp,
               int M, int N, int K, int ldb, int ldcT, float alpha,
               long batchA, long batchB, long batchC)
{
  const int tid  = threadIdx.x;
  const int lane = tid & 31;
  const int wave = tid >> 5;
  const int wm   = wave >> 2;   // 0..1 : 64-row slab
  const int wn   = wave & 3;    // 0..3 : 32-col slab
  const int lr   = lane & 15;
  const int lh   = lane >> 4;

  const int m0 = blockIdx.y * BM;
  const int n0 = blockIdx.x * BN;
  const int bz = blockIdx.z;

  const bf16_t* Ab = Ap + (size_t)bz * batchA;
  const bf16_t* Bb = Bp + (size_t)bz * batchB;

  __shared__ __attribute__((aligned(128))) bf16_t As[2][BM * BK]; // [m][k]
  __shared__ __attribute__((aligned(128))) bf16_t Bs[2][BN * BK]; // [n][k]

  // ---- per-thread async-copy slots: 4 x b128 for A, 4 x b128 for B -------
  const uint32_t asBase = (uint32_t)(uintptr_t)&As[0][0];
  const uint32_t bsBase = (uint32_t)(uintptr_t)&Bs[0][0];
  const bf16_t* aG[4];
  const bf16_t* bG[4];
  uint32_t aOff[4];              // byte offset inside one LDS buffer
#pragma unroll
  for (int i = 0; i < 4; ++i) {
    int linear = (tid + i * 256) * 8;      // 8192 bf16 per tile
    int mm = linear >> 6, kk = linear & 63;
    aG[i]   = Ab + (size_t)(m0 + mm) * K + kk;
    bG[i]   = Bb + (size_t)(n0 + mm) * ldb + kk;   // same decomposition
    aOff[i] = (uint32_t)((mm * BK + kk) * 2);
  }

  auto issue_tile = [&](int buf, int koff) {
#pragma unroll
    for (int i = 0; i < 4; ++i) {
      uint32_t la = asBase + (uint32_t)(buf * (BM * BK * 2)) + aOff[i];
      const void* ga = (const void*)(aG[i] + koff);
      asm volatile("global_load_async_to_lds_b128 %0, %1, off"
                   :: "v"(la), "v"(ga) : "memory");
    }
#pragma unroll
    for (int i = 0; i < 4; ++i) {
      uint32_t lb = bsBase + (uint32_t)(buf * (BN * BK * 2)) + aOff[i];
      const void* gb = (const void*)(bG[i] + koff);
      asm volatile("global_load_async_to_lds_b128 %0, %1, off"
                   :: "v"(lb), "v"(gb) : "memory");
    }
  };

  v8f zero;
#pragma unroll
  for (int i = 0; i < 8; ++i) zero[i] = 0.0f;
  v8f acc[4][2];
#pragma unroll
  for (int mt = 0; mt < 4; ++mt)
#pragma unroll
    for (int nt = 0; nt < 2; ++nt) acc[mt][nt] = zero;

  // 2 K-substeps of 32 each, 8 WMMAs per substep, on buffer `buf`.
  auto compute_tile = [&](int buf) {
#pragma unroll
    for (int ks = 0; ks < 2; ++ks) {
      // A 16x32 bf16 frag: lanes 0-15 row=lr K 0..15,
      // lanes 16-31 row=lr K 16..31 -> one contiguous v16bf read.
      v16bf afrag[4];
#pragma unroll
      for (int mt = 0; mt < 4; ++mt) {
        int row = wm * 64 + mt * 16 + lr;
        afrag[mt] = *(const v16bf*)&As[buf][row * BK + ks * 32 + lh * 16];
      }
      // B 32x16 bf16 frag: lanes 0-15 col=lr K 0..15, lanes 16-31 K 16..31
      // -> one contiguous 32B read from the [n][k] tile.
      v16bf bfrag[2];
#pragma unroll
      for (int nt = 0; nt < 2; ++nt) {
        int col = wn * 32 + nt * 16 + lr;
        bfrag[nt] = *(const v16bf*)&Bs[buf][col * BK + ks * 32 + lh * 16];
      }
#pragma unroll
      for (int mt = 0; mt < 4; ++mt)
#pragma unroll
        for (int nt = 0; nt < 2; ++nt)
          acc[mt][nt] = __builtin_amdgcn_wmma_f32_16x16x32_bf16(
              false, afrag[mt], false, bfrag[nt],
              (short)0, acc[mt][nt], false, false);
    }
  };

  const int nsteps = K / BK;     // >= 2 for all uses (K = 1024 or 2048)
  issue_tile(0, 0);              // prologue: tile 0 -> buffer 0
  int cur = 0;

  // Steady state (branch-free): stream tile s+1 while computing tile s.
  for (int s = 0; s < nsteps - 1; ++s) {
    __syncthreads();                       // buf[cur^1] reads from s-1 done
    issue_tile(cur ^ 1, (s + 1) * BK);     // 8 async ops per wave
    asm volatile("s_wait_asynccnt 0x8" ::: "memory");  // oldest 8 landed
    __syncthreads();                       // buf[cur] visible to all waves
    compute_tile(cur);
    cur ^= 1;
  }
  // Final tile: nothing left to stream.
  __syncthreads();
  asm volatile("s_wait_asynccnt 0x0" ::: "memory");
  __syncthreads();
  compute_tile(cur);

  // ---- epilogue: C layout VGPR r -> row r (lanes 0-15) / r+8 (16-31) -----
  float*  Cf = (float*)Cp  + (size_t)bz * batchC;
  bf16_t* Cb = (bf16_t*)Cp + (size_t)bz * batchC;
#pragma unroll
  for (int mt = 0; mt < 4; ++mt) {
#pragma unroll
    for (int nt = 0; nt < 2; ++nt) {
      int gm = m0 + wm * 64 + mt * 16 + lh * 8;
      int gn = n0 + wn * 32 + nt * 16 + lr;
      float bv = BIAS ? bias[gn] : 0.0f;
      if (OUT_TRANS) {
        // transposed store: 8 consecutive rows are contiguous -> one b128
        union { v8bf v; bf16_t e[8]; } p;
#pragma unroll
        for (int r = 0; r < 8; ++r)
          p.e[r] = (bf16_t)(acc[mt][nt][r] * alpha + bv);
        *(v8bf*)(Cb + (size_t)gn * ldcT + gm) = p.v;
      } else {
#pragma unroll
        for (int r = 0; r < 8; ++r) {
          float val = acc[mt][nt][r] * alpha + bv;
          size_t idx = (size_t)(gm + r) * N + gn;
          if (OUT_F32) Cf[idx] = val;
          else         Cb[idx] = (bf16_t)val;
        }
      }
    }
  }
}

// ---------------------------------------------------------------------------
// Row softmax over 2048 columns; one 256-thread block per row.
// ---------------------------------------------------------------------------
__global__ __launch_bounds__(256)
void softmax_rows(const float* __restrict__ S, bf16_t* __restrict__ P)
{
  const int cols = SEQ;
  const size_t row = blockIdx.x;
  const float* src = S + row * (size_t)cols;
  bf16_t*      dst = P + row * (size_t)cols;
  const int tid = threadIdx.x;

  float v[8];
  float lmax = -3.4e38f;
#pragma unroll
  for (int i = 0; i < 8; ++i) {
    v[i] = src[tid + i * 256];
    lmax = fmaxf(lmax, v[i]);
  }

  __shared__ float redm[8];
  __shared__ float reds[8];
  for (int off = 16; off > 0; off >>= 1)
    lmax = fmaxf(lmax, __shfl_xor(lmax, off, 32));
  if ((tid & 31) == 0) redm[tid >> 5] = lmax;
  __syncthreads();
  float rmax = redm[0];
#pragma unroll
  for (int i = 1; i < 8; ++i) rmax = fmaxf(rmax, redm[i]);

  float lsum = 0.0f;
#pragma unroll
  for (int i = 0; i < 8; ++i) {
    v[i] = __expf(v[i] - rmax);
    lsum += v[i];
  }
  for (int off = 16; off > 0; off >>= 1)
    lsum += __shfl_xor(lsum, off, 32);
  if ((tid & 31) == 0) reds[tid >> 5] = lsum;
  __syncthreads();
  float rsum = 0.0f;
#pragma unroll
  for (int i = 0; i < 8; ++i) rsum += reds[i];
  float inv = 1.0f / rsum;
#pragma unroll
  for (int i = 0; i < 8; ++i)
    dst[tid + i * 256] = (bf16_t)(v[i] * inv);
}

// ---------------------------------------------------------------------------
// Elementwise fp32 -> bf16.
// ---------------------------------------------------------------------------
__global__ __launch_bounds__(256)
void cvt_f32_bf16(const float* __restrict__ x, bf16_t* __restrict__ y, int n)
{
  int i = (blockIdx.x * 256 + threadIdx.x) * 4;
  if (i + 3 < n) {
    float4 a = *(const float4*)(x + i);
    y[i]     = (bf16_t)a.x;
    y[i + 1] = (bf16_t)a.y;
    y[i + 2] = (bf16_t)a.z;
    y[i + 3] = (bf16_t)a.w;
  }
}

// ---------------------------------------------------------------------------
extern "C" void kernel_launch(void* const* d_in, const int* in_sizes, int n_in,
                              void* d_out, int out_size, void* d_ws, size_t ws_size,
                              hipStream_t stream)
{
  const float* q  = (const float*)d_in[0];
  const float* k  = (const float*)d_in[1];
  const float* v  = (const float*)d_in[2];
  const float* Wq = (const float*)d_in[3];
  const float* bq = (const float*)d_in[4];
  const float* Wk = (const float*)d_in[5];
  const float* bk = (const float*)d_in[6];
  const float* Wv = (const float*)d_in[7];
  const float* bv = (const float*)d_in[8];
  const float* Wo = (const float*)d_in[9];
  const float* bo = (const float*)d_in[10];
  float* out = (float*)d_out;

  // ---- workspace layout (bytes) ----
  char* ws = (char*)d_ws;
  const size_t MB = 1ull << 20;
  bf16_t* Wqb = (bf16_t*)(ws + 0 * MB);    //  2 MB
  bf16_t* Wkb = (bf16_t*)(ws + 2 * MB);    //  2 MB
  bf16_t* Wvb = (bf16_t*)(ws + 4 * MB);    //  2 MB
  bf16_t* Wob = (bf16_t*)(ws + 6 * MB);    //  2 MB
  bf16_t* qb  = (bf16_t*)(ws + 8 * MB);    // 16 MB  q in bf16
  bf16_t* kb  = (bf16_t*)(ws + 24 * MB);   // 16 MB
  bf16_t* vb  = (bf16_t*)(ws + 40 * MB);   // 16 MB
  bf16_t* qp  = (bf16_t*)(ws + 56 * MB);   // 16 MB  [B*S, D]
  bf16_t* kp  = (bf16_t*)(ws + 72 * MB);   // 16 MB  [B*S, D]
  bf16_t* vpT = (bf16_t*)(ws + 88 * MB);   // 16 MB  [D, B*S]  (transposed!)
  bf16_t* ctx = (bf16_t*)(ws + 104 * MB);  // 16 MB  [B*S, D]
  float*  sc  = (float*)(ws + 120 * MB);   // 64 MB  [B,S,S] fp32
  bf16_t* att = (bf16_t*)(ws + 184 * MB);  // 32 MB  [B,S,S] bf16

  const int DD    = D_MODEL * D_MODEL;     // 1M
  const int Mrows = BATCH * SEQ;           // 8192
  const int NX    = Mrows * D_MODEL;       // 8.4M

  // 1) weights + activations -> bf16
  cvt_f32_bf16<<<dim3(DD / 1024), 256, 0, stream>>>(Wq, Wqb, DD);
  cvt_f32_bf16<<<dim3(DD / 1024), 256, 0, stream>>>(Wk, Wkb, DD);
  cvt_f32_bf16<<<dim3(DD / 1024), 256, 0, stream>>>(Wv, Wvb, DD);
  cvt_f32_bf16<<<dim3(DD / 1024), 256, 0, stream>>>(Wo, Wob, DD);
  cvt_f32_bf16<<<dim3(NX / 1024), 256, 0, stream>>>(q, qb, NX);
  cvt_f32_bf16<<<dim3(NX / 1024), 256, 0, stream>>>(k, kb, NX);
  cvt_f32_bf16<<<dim3(NX / 1024), 256, 0, stream>>>(v, vb, NX);

  // 2) projections: [8192,1024] x [1024,1024]^T + bias
  dim3 gP(D_MODEL / BN, Mrows / BM, 1);    // (8, 64, 1)
  gemm_wmma<false, true, false><<<gP, 256, 0, stream>>>(
      qb, Wqb, bq, qp, Mrows, D_MODEL, D_MODEL, D_MODEL, 0, 1.0f, 0, 0, 0);
  gemm_wmma<false, true, false><<<gP, 256, 0, stream>>>(
      kb, Wkb, bk, kp, Mrows, D_MODEL, D_MODEL, D_MODEL, 0, 1.0f, 0, 0, 0);
  // v projection written transposed: vpT[d][m], row stride 8192
  gemm_wmma<false, true, true><<<gP, 256, 0, stream>>>(
      vb, Wvb, bv, vpT, Mrows, D_MODEL, D_MODEL, D_MODEL, Mrows, 1.0f, 0, 0, 0);

  // 3) scores = qp kp^T * 1/sqrt(64), batched over B
  dim3 gS(SEQ / BN, SEQ / BM, BATCH);      // (16, 16, 4)
  gemm_wmma<true, false, false><<<gS, 256, 0, stream>>>(
      qp, kp, nullptr, sc, SEQ, SEQ, D_MODEL, D_MODEL, 0, 0.125f,
      (long)SEQ * D_MODEL, (long)SEQ * D_MODEL, (long)SEQ * SEQ);

  // 4) softmax rows (8192 rows of 2048), fp32 -> bf16
  softmax_rows<<<dim3(Mrows), 256, 0, stream>>>(sc, att);

  // 5) ctx = att * vp  ==  att * (vpT slice)^T : NT GEMM, ldb = 8192,
  //    per-batch B offset = SEQ columns into vpT
  dim3 gC(D_MODEL / BN, SEQ / BM, BATCH);  // (8, 16, 4)
  gemm_wmma<false, false, false><<<gC, 256, 0, stream>>>(
      att, vpT, nullptr, ctx, SEQ, D_MODEL, SEQ, Mrows, 0, 1.0f,
      (long)SEQ * SEQ, (long)SEQ, (long)SEQ * D_MODEL);

  // 6) out = ctx Wo^T + bo -> fp32 d_out
  gemm_wmma<true, true, false><<<gP, 256, 0, stream>>>(
      ctx, Wob, bo, out, Mrows, D_MODEL, D_MODEL, D_MODEL, 0, 1.0f, 0, 0, 0);
}
